// Similarity_Search_78623671320889
// MI455X (gfx1250) — matrix operand order repaired and checked by reading, not compiled, not run on hardware
//
#include <hip/hip_runtime.h>
#include <hip/hip_bf16.h>

typedef __attribute__((ext_vector_type(16))) _Float16 v16h;
typedef __attribute__((ext_vector_type(8)))  float    v8f;

#define D_Q        256
#define C_DIM      256
#define N_DB       500000
#define ROW_STRIDE 257        // places_db row: 256 floats + class id
#define TOPK       10
#define MIN_SIM    0.8f
#define OV_THR     0.8f
#define CAND_CAP   1024
#define NB         80         // db rows per workgroup (5 n-tiles of 16); 500000 = 80*6250

// ---------------- workspace layout (bytes) ----------------
// afrag      : 65536 halves (A fragments, WMMA lane order)   @ 0        (131072 B)
// cand_count : 256 int                                       @ 131072
// cand_sim   : 256*1024 float                                @ 132096
// cand_id    : 256*1024 int                                  @ 1180672
// topk_sim   : 256*10 float                                  @ 2229248
// topk_id    : 256*10 int                                    @ 2239488
// results_pre: 256 int                                       @ 2249728
#define WS_AFRAG   0
#define WS_CCNT    131072
#define WS_CSIM    132096
#define WS_CID     1180672
#define WS_TSIM    2229248
#define WS_TID     2239488
#define WS_RES     2249728

// A-fragment half index <-> K mapping (16-bit A 16x32 layout, ISA 7.12.2):
//   lanes 0-15 hold K {0..7,16..23}, lanes 16-31 hold K {8..15,24..31}
//   half idx 0..7  -> kk = idx + 8*hi
//   half idx 8..15 -> kk = idx + 8 + 8*hi
// B-fragment (K x 16): same K order as A (lanes 0-15 hold K {0..7,16..23}, lanes 16-31 hold K {8..15,24..31}).

__global__ __launch_bounds__(256) void simsearch_pack_a(const float* __restrict__ desc,
                                                        _Float16* __restrict__ afrag,
                                                        int* __restrict__ cand_count) {
    int gid = blockIdx.x * 256 + threadIdx.x;       // [0, 65536)
    int idx  = gid & 15;
    int lane = (gid >> 4) & 31;
    int c    = (gid >> 9) & 7;
    int t    = gid >> 12;                            // m-tile 0..15
    int m  = t * 16 + (lane & 15);
    int hi = lane >> 4;
    int kk = (idx & 7) + (hi << 3) + ((idx >> 3) << 4);
    int k  = c * 32 + kk;
    afrag[gid] = (_Float16)desc[m * C_DIM + k];
    if (gid < D_Q) cand_count[gid] = 0;
}

__global__ __launch_bounds__(256) void simsearch_gemm(const float* __restrict__ pdb,
                                                      const _Float16* __restrict__ afrag,
                                                      int* __restrict__ cand_count,
                                                      float* __restrict__ cand_sim,
                                                      int* __restrict__ cand_id) {
    // B fragments in LDS: [ntile(5)][chunk(8)][lane(32)][half(16)] = 20480 halves = 40 KB
    __shared__ __attribute__((aligned(32))) _Float16 ldsB[5 * 8 * 32 * 16];

    const int tid   = threadIdx.x;
    const int nbase = blockIdx.x * NB;

    // ---- cooperative pack of 80 db rows into B-fragment order ----
    {
        const int k  = tid;            // each thread owns one K column (0..255)
        const int c  = k >> 5;
        const int kk = k & 31;
        const int hi = (kk >> 3) & 1;
        const int hidx = (kk & 7) + ((kk >> 4) << 3);
        #pragma unroll 8
        for (int i = 0; i < NB; ++i) {
            float v = pdb[(size_t)(nbase + i) * ROW_STRIDE + k];
            int ntile = i >> 4;
            int blane = (i & 15) + (hi << 4);
            ldsB[(((ntile * 8 + c) * 32 + blane) << 4) + hidx] = (_Float16)v;
        }
    }
    __syncthreads();

    const int lane = tid & 31;
    const int wave = tid >> 5;
    const int t0 = wave * 2, t1 = wave * 2 + 1;

    v8f acc[10];
    #pragma unroll
    for (int i = 0; i < 10; ++i) acc[i] = (v8f){};

    #pragma unroll
    for (int c = 0; c < 8; ++c) {
        v16h a0 = *(const v16h*)(afrag + (((t0 * 8 + c) * 32 + lane) << 4));
        v16h a1 = *(const v16h*)(afrag + (((t1 * 8 + c) * 32 + lane) << 4));
        #pragma unroll
        for (int nt = 0; nt < 5; ++nt) {
            v16h b = *(const v16h*)(&ldsB[((nt * 8 + c) * 32 + lane) << 4]);
            acc[2 * nt]     = __builtin_amdgcn_wmma_f32_16x16x32_f16(
                false, a0, false, b, (short)0, acc[2 * nt],     false, false);
            acc[2 * nt + 1] = __builtin_amdgcn_wmma_f32_16x16x32_f16(
                false, a1, false, b, (short)0, acc[2 * nt + 1], false, false);
        }
    }

    // ---- harvest candidates >= MIN_SIM ----
    const int mrow_off = (lane >> 4) * 8;      // C/D layout: VGPR r -> M = r (+8 for hi lanes)
    const int ncol_loc = lane & 15;
    #pragma unroll
    for (int nt = 0; nt < 5; ++nt) {
        int n = nbase + nt * 16 + ncol_loc;
        float cls = pdb[(size_t)n * ROW_STRIDE + C_DIM];
        #pragma unroll
        for (int r = 0; r < 8; ++r) {
            float s0 = acc[2 * nt][r];
            if (s0 >= MIN_SIM) {
                int m = t0 * 16 + r + mrow_off;
                int pos = atomicAdd(&cand_count[m], 1);
                if (pos < CAND_CAP) {
                    cand_sim[m * CAND_CAP + pos] = s0;
                    cand_id [m * CAND_CAP + pos] = (int)cls;
                }
            }
            float s1 = acc[2 * nt + 1][r];
            if (s1 >= MIN_SIM) {
                int m = t1 * 16 + r + mrow_off;
                int pos = atomicAdd(&cand_count[m], 1);
                if (pos < CAND_CAP) {
                    cand_sim[m * CAND_CAP + pos] = s1;
                    cand_id [m * CAND_CAP + pos] = (int)cls;
                }
            }
        }
    }
}

__global__ __launch_bounds__(256) void simsearch_vote(const int* __restrict__ cand_count,
                                                      const float* __restrict__ cand_sim,
                                                      const int* __restrict__ cand_id,
                                                      float* __restrict__ topk_sim,
                                                      int* __restrict__ topk_id,
                                                      int* __restrict__ results_pre) {
    int m = threadIdx.x;
    int cnt = cand_count[m];
    if (cnt > CAND_CAP) cnt = CAND_CAP;

    float ts[TOPK]; int ti[TOPK];
    #pragma unroll
    for (int i = 0; i < TOPK; ++i) { ts[i] = -1e30f; ti[i] = -2; }

    for (int i = 0; i < cnt; ++i) {
        float s = cand_sim[m * CAND_CAP + i];
        int   id = cand_id[m * CAND_CAP + i];
        if (s > ts[TOPK - 1]) {
            int p = TOPK - 1;
            #pragma unroll
            for (int q = TOPK - 1; q > 0; --q) {
                if (ts[q - 1] < s) { ts[q] = ts[q - 1]; ti[q] = ti[q - 1]; p = q - 1; }
            }
            ts[p] = s; ti[p] = id;
        }
    }
    int kvalid = cnt < TOPK ? cnt : TOPK;

    // majority vote: max count, smallest class id on ties (jnp.argmax semantics)
    int bestId = -1, bestCnt = 0;
    #pragma unroll
    for (int i = 0; i < TOPK; ++i) {
        if (i >= kvalid) break;
        int c = 0;
        #pragma unroll
        for (int j = 0; j < TOPK; ++j)
            if (j < kvalid && ti[j] == ti[i]) ++c;
        if (c > bestCnt || (c == bestCnt && ti[i] < bestId)) { bestCnt = c; bestId = ti[i]; }
    }
    results_pre[m] = (bestCnt > 0) ? bestId : -1;

    #pragma unroll
    for (int i = 0; i < TOPK; ++i) {
        bool v = (i < kvalid);
        topk_sim[m * TOPK + i] = v ? ts[i] : 0.0f;
        topk_id [m * TOPK + i] = v ? ti[i] : -2;
    }
}

__global__ __launch_bounds__(256) void simsearch_nms_out(const float* __restrict__ boxes,
                                                         const int* __restrict__ results_pre,
                                                         const float* __restrict__ topk_sim,
                                                         const int* __restrict__ topk_id,
                                                         float* __restrict__ out) {
    __shared__ float bx[D_Q][4];
    __shared__ float sarea[D_Q];
    __shared__ int   slab[D_Q];
    int i = threadIdx.x;
    float x1 = boxes[i * 4 + 0], y1 = boxes[i * 4 + 1];
    float x2 = boxes[i * 4 + 2], y2 = boxes[i * 4 + 3];
    bx[i][0] = x1; bx[i][1] = y1; bx[i][2] = x2; bx[i][3] = y2;
    float ai = (x2 - x1) * (y2 - y1);
    sarea[i] = ai;
    slab[i] = results_pre[i];
    __syncthreads();

    int li = slab[i];
    bool remove = false;
    if (li >= 0) {
        // remove[i] <=> exists j!=i : same label, overlap >= thr, area_j <= area_i
        for (int j = 0; j < D_Q; ++j) {
            if (j == i || slab[j] != li) continue;
            float aj = sarea[j];
            if (aj > ai) continue;
            float xi1 = fmaxf(x1, bx[j][0]);
            float yi1 = fmaxf(y1, bx[j][1]);
            float xi2 = fminf(x2, bx[j][2]);
            float yi2 = fminf(y2, bx[j][3]);
            float inter = fmaxf(xi2 - xi1, 0.0f) * fmaxf(yi2 - yi1, 0.0f);
            float asmall = fminf(ai, aj);
            float ov = (asmall > 0.0f) ? (inter / asmall) : 0.0f;
            if (ov >= OV_THR) { remove = true; break; }
        }
    }
    int res = remove ? -1 : li;

    float sc = 0.0f;
    #pragma unroll
    for (int t = 0; t < TOPK; ++t) {
        if (topk_id[i * TOPK + t] == res)
            sc = fmaxf(sc, topk_sim[i * TOPK + t]);
    }

    // outputs flat: boxes (1024) | sim_scores (256) | results (256, as float)
    out[i * 4 + 0] = x1; out[i * 4 + 1] = y1;
    out[i * 4 + 2] = x2; out[i * 4 + 3] = y2;
    out[1024 + i] = sc;
    out[1280 + i] = (float)res;
}

extern "C" void kernel_launch(void* const* d_in, const int* in_sizes, int n_in,
                              void* d_out, int out_size, void* d_ws, size_t ws_size,
                              hipStream_t stream) {
    const float* boxes = (const float*)d_in[0];   // (256,4) f32
    const float* desc  = (const float*)d_in[1];   // (256,256) f32
    const float* pdb   = (const float*)d_in[2];   // (500000,257) f32
    float* out = (float*)d_out;

    char* ws = (char*)d_ws;                       // needs ~2.25 MB
    _Float16* afrag     = (_Float16*)(ws + WS_AFRAG);
    int*      cand_cnt  = (int*)     (ws + WS_CCNT);
    float*    cand_sim  = (float*)   (ws + WS_CSIM);
    int*      cand_id   = (int*)     (ws + WS_CID);
    float*    topk_sim  = (float*)   (ws + WS_TSIM);
    int*      topk_id   = (int*)     (ws + WS_TID);
    int*      res_pre   = (int*)     (ws + WS_RES);

    simsearch_pack_a<<<256, 256, 0, stream>>>(desc, afrag, cand_cnt);
    simsearch_gemm<<<N_DB / NB, 256, 0, stream>>>(pdb, afrag, cand_cnt, cand_sim, cand_id);
    simsearch_vote<<<1, 256, 0, stream>>>(cand_cnt, cand_sim, cand_id, topk_sim, topk_id, res_pre);
    simsearch_nms_out<<<1, 256, 0, stream>>>(boxes, res_pre, topk_sim, topk_id, out);
}
